// FixedVectorQuantizer_87041807220937
// MI455X (gfx1250) — matrix operand order repaired and checked by reading, not compiled, not run on hardware
//
#include <hip/hip_runtime.h>
#include <hip/hip_bf16.h>

typedef __bf16 bf16;
typedef __attribute__((ext_vector_type(16))) __bf16 v16bf;
typedef __attribute__((ext_vector_type(8)))  __bf16 v8bf;
typedef __attribute__((ext_vector_type(4)))  __bf16 v4bf;
typedef __attribute__((ext_vector_type(8)))  float  v8f;

#define N_ROWS 32768
#define D_DIM  1024
#define K_DIM  4096
#define MB     64                       // rows per workgroup in search kernel
#define ROW_BYTES   2048                // global row: 1024 bf16
#define LROW_BYTES  2064                // padded LDS row (+16B -> bank-conflict-free)
#define A_BYTES     (MB * LROW_BYTES)   // 132096
#define B_BUF_BYTES (32 * LROW_BYTES)   //  66048 per buffer
#define SMEM_MAIN   (A_BYTES + 2 * B_BUF_BYTES)   // 264192 bytes dynamic LDS

__device__ __forceinline__ void wait_async0() {
#if __has_builtin(__builtin_amdgcn_s_wait_asynccnt)
  __builtin_amdgcn_s_wait_asynccnt(0);
#else
  asm volatile("s_wait_asynccnt 0x0" ::: "memory");
#endif
}

// One 16B async global->LDS copy (ASYNCcnt-tracked). loff = byte offset into
// workgroup LDS, goff = byte offset from base, base = global address (SGPR pair).
__device__ __forceinline__ void async_cp16(unsigned loff, unsigned goff,
                                           unsigned long long base) {
  asm volatile("global_load_async_to_lds_b128 %0, %1, %2"
               :: "v"(loff), "v"(goff), "s"(base) : "memory");
}

// Stage `rows` padded rows (2048B each) from row-major bf16 src into LDS.
__device__ __forceinline__ void stage_rows_async(const bf16* src, int rows,
                                                 unsigned lds_base, int tid) {
  const unsigned long long base = (unsigned long long)(uintptr_t)src;
  const int nchunk = rows * (ROW_BYTES / 16);
  for (int c = tid; c < nchunk; c += 256) {
    const int r   = c >> 7;            // 128 x 16B chunks per row
    const int col = (c & 127) << 4;
    async_cp16(lds_base + r * LROW_BYTES + col,
               (unsigned)(r * ROW_BYTES + col), base);
  }
}

__device__ __forceinline__ v16bf pack16(v8bf lo, v8bf hi) {
  v16bf r;
  #pragma unroll
  for (int j = 0; j < 8; ++j) { r[j] = lo[j]; r[j + 8] = hi[j]; }
  return r;
}

struct Frag { v8bf al, ah, bl, bh; };

__device__ __forceinline__ Frag ld_frag(const bf16* A, const bf16* B, int d0) {
  Frag f;
  f.al = *(const v8bf*)(A + d0);
  f.ah = *(const v8bf*)(A + d0 + 16);
  f.bl = *(const v8bf*)(B + d0);
  f.bh = *(const v8bf*)(B + d0 + 16);
  return f;
}

__device__ __forceinline__ v8f do_wmma(const Frag& f, v8f c) {
  return __builtin_amdgcn_wmma_f32_16x16x32_bf16(
      false, pack16(f.al, f.ah), false, pack16(f.bl, f.bh), (short)0, c,
      false, false);
}

// ---------------------------------------------------------------------------
// Kernel 1: fp32 row -> bf16 row + fp32 squared norm. One block per row.
// ---------------------------------------------------------------------------
__global__ void __launch_bounds__(256)
vq_convert_rows(const float* __restrict__ src, bf16* __restrict__ dst,
                float* __restrict__ norms) {
  const int row = blockIdx.x;
  const int tid = threadIdx.x;
  const size_t base = (size_t)row * D_DIM + tid * 4;
  const float4 v = *(const float4*)(src + base);
  v4bf b;
  b[0] = (bf16)v.x; b[1] = (bf16)v.y; b[2] = (bf16)v.z; b[3] = (bf16)v.w;
  *(v4bf*)(dst + base) = b;
  float s = v.x * v.x + v.y * v.y + v.z * v.z + v.w * v.w;

  __shared__ float red[256];
  red[tid] = s;
  __syncthreads();
  #pragma unroll
  for (int off = 128; off > 0; off >>= 1) {
    if (tid < off) red[tid] += red[tid + off];
    __syncthreads();
  }
  if (tid == 0) norms[row] = red[0];
}

// ---------------------------------------------------------------------------
// Kernel 2: fused distance search. 64 rows/block, 8 waves (4 row-tiles x
// 2 code-tiles). A block pinned in LDS; 32-code B tiles double-buffered in
// LDS via async global->LDS DMA overlapping the WMMA stream. Inner loop
// issues one 16-load batch per 128 dims, consumed by 4 WMMAs on 4
// independent accumulators (no loop-carried fragment tuples -> no copies).
// Branchless online min/argmin + logsumexp over all 4096 codes.
// ---------------------------------------------------------------------------
__global__ void __launch_bounds__(256)
vq_search(const bf16* __restrict__ Abf, const bf16* __restrict__ Cbf,
          const float* __restrict__ x2, const float* __restrict__ c2,
          float* __restrict__ min_out, float* __restrict__ idxf_out,
          int* __restrict__ idxi_out, float* __restrict__ rowcls) {
  const int tid  = threadIdx.x;
  const int lane = tid & 31;
  const int wid  = tid >> 5;
  const int wm   = wid & 3;            // row-tile (0..3)
  const int wn   = wid >> 2;           // code-tile within 32-code step (0..1)
  const int rowBase = blockIdx.x * MB;

  extern __shared__ char smem[];

  // async-stage the 64x1024 A block and the first B tile
  stage_rows_async(Abf + (size_t)rowBase * D_DIM, MB, 0u, tid);
  stage_rows_async(Cbf, 32, A_BYTES, tid);

  const int halfSel = lane >> 4;
  const int l15     = lane & 15;

  float rx2[8];
  #pragma unroll
  for (int i = 0; i < 8; ++i)
    rx2[i] = x2[rowBase + wm * 16 + i + 8 * halfSel];

  float minv[8], mlse[8], slse[8];
  int   amin[8];
  #pragma unroll
  for (int i = 0; i < 8; ++i) {
    minv[i] = 3.4e38f; mlse[i] = -3.4e38f; slse[i] = 0.0f; amin[i] = 0;
  }

  const bf16* sArow =
      (const bf16*)(smem + (size_t)(wm * 16 + l15) * LROW_BYTES + halfSel * 16);

  int buf = 0;
  for (int ct = 0; ct < K_DIM; ct += 32, buf ^= 1) {
    // my async DMA into `buf` complete; barrier makes it visible block-wide
    // and guarantees everyone is done reading buf^1 (their previous compute)
    wait_async0();
    __syncthreads();
    // prefetch next 32-code tile into the other buffer while we compute
    if (ct + 32 < K_DIM)
      stage_rows_async(Cbf + (size_t)(ct + 32) * D_DIM, 32,
                       (unsigned)(A_BYTES + (buf ^ 1) * B_BUF_BYTES), tid);

    const int code = ct + wn * 16 + l15;
    const float c2v = c2[code];
    const bf16* sBrow = (const bf16*)(smem + A_BYTES + buf * B_BUF_BYTES +
                                      (size_t)(wn * 16 + l15) * LROW_BYTES +
                                      halfSel * 16);

    // batch-of-4 inner product: 16 DS loads then 4 WMMAs per 128 dims;
    // fragment sets die inside the iteration, accumulators stay in place
    v8f acc0 = {}, acc1 = {}, acc2 = {}, acc3 = {};
    #pragma unroll 1
    for (int d0 = 0; d0 < D_DIM; d0 += 128) {
      const Frag g0 = ld_frag(sArow, sBrow, d0);
      const Frag g1 = ld_frag(sArow, sBrow, d0 + 32);
      const Frag g2 = ld_frag(sArow, sBrow, d0 + 64);
      const Frag g3 = ld_frag(sArow, sBrow, d0 + 96);
      acc0 = do_wmma(g0, acc0);
      acc1 = do_wmma(g1, acc1);
      acc2 = do_wmma(g2, acc2);
      acc3 = do_wmma(g3, acc3);
    }

    // branchless distance + online min / logsumexp(-d)
    #pragma unroll
    for (int i = 0; i < 8; ++i) {
      const float dot = (acc0[i] + acc1[i]) + (acc2[i] + acc3[i]);
      const float dist = rx2[i] + c2v - 2.0f * dot;
      const bool better = dist < minv[i];
      minv[i] = better ? dist : minv[i];
      amin[i] = better ? code : amin[i];
      const float nd   = -dist;
      const float newm = fmaxf(mlse[i], nd);
      slse[i] = slse[i] * __expf(mlse[i] - newm) + __expf(nd - newm);
      mlse[i] = newm;
    }
  }

  // --- cross-lane/wave reduction through LDS (A/B regions reused) ---
  __syncthreads();
  float* sred = (float*)smem;          // [MB][32 slots][4 floats]
  const int slot = wn * 16 + l15;
  #pragma unroll
  for (int i = 0; i < 8; ++i) {
    const int rl = wm * 16 + i + 8 * halfSel;
    float* p = sred + ((rl * 32 + slot) << 2);
    p[0] = minv[i];
    p[1] = __int_as_float(amin[i]);
    p[2] = mlse[i];
    p[3] = slse[i];
  }
  __syncthreads();

  if (tid < MB) {
    const float* base = sred + (tid * 32) * 4;
    float bm = 3.4e38f; int bi = 0x7fffffff;
    float M = -3.4e38f; float S = 0.0f;
    for (int s = 0; s < 32; ++s) {
      const float* p = base + s * 4;
      const float dv = p[0];
      const int   iv = __float_as_int(p[1]);
      const bool better = (dv < bm) || (dv == bm && iv < bi);
      bm = better ? dv : bm;
      bi = better ? iv : bi;
      const float m2 = p[2], s2 = p[3];
      const float nM = fmaxf(M, m2);
      S = S * __expf(M - nM) + s2 * __expf(m2 - nM);
      M = nM;
    }
    const int n = rowBase + tid;
    min_out[n]  = bm;
    idxf_out[n] = (float)bi;
    idxi_out[n] = bi;
    rowcls[n]   = bm + M + __logf(S);   // d_min + logsumexp(-d) = NLL term
  }
}

// ---------------------------------------------------------------------------
// Kernel 3: gather codebook rows (straight-through output) + exact fp32
// commitment partial sums. One block per row.
// ---------------------------------------------------------------------------
__global__ void __launch_bounds__(256)
vq_gather(const int* __restrict__ idx, const float* __restrict__ cb,
          const float* __restrict__ inputs, float* __restrict__ outq,
          float* __restrict__ commit_partial) {
  const int row = blockIdx.x;
  const int tid = threadIdx.x;
  const int k = idx[row];
  const size_t jo = (size_t)tid * 4;
  const float4 c = *(const float4*)(cb + (size_t)k * D_DIM + jo);
  const float4 x = *(const float4*)(inputs + (size_t)row * D_DIM + jo);
  *(float4*)(outq + (size_t)row * D_DIM + jo) = c;
  const float dx = x.x - c.x, dy = x.y - c.y, dz = x.z - c.z, dw = x.w - c.w;
  float s = dx * dx + dy * dy + dz * dz + dw * dw;

  __shared__ float red[256];
  red[tid] = s;
  __syncthreads();
  #pragma unroll
  for (int off = 128; off > 0; off >>= 1) {
    if (tid < off) red[tid] += red[tid + off];
    __syncthreads();
  }
  if (tid == 0) commit_partial[row] = red[0];
}

// ---------------------------------------------------------------------------
// Kernel 4: deterministic final reduction for the two scalar losses.
// ---------------------------------------------------------------------------
__global__ void __launch_bounds__(256)
vq_finalize(const float* __restrict__ rowcls, const float* __restrict__ cpart,
            float* __restrict__ out_commit, float* __restrict__ out_cls) {
  __shared__ float r1[256], r2[256];
  const int tid = threadIdx.x;
  float a = 0.0f, b = 0.0f;
  for (int i = tid; i < N_ROWS; i += 256) { a += rowcls[i]; b += cpart[i]; }
  r1[tid] = a; r2[tid] = b;
  __syncthreads();
  #pragma unroll
  for (int off = 128; off > 0; off >>= 1) {
    if (tid < off) { r1[tid] += r1[tid + off]; r2[tid] += r2[tid + off]; }
    __syncthreads();
  }
  if (tid == 0) {
    out_cls[0]    = r1[0] / (float)N_ROWS;
    out_commit[0] = r2[0] * 0.25f;
  }
}

// ---------------------------------------------------------------------------
static inline size_t align256(size_t x) { return (x + 255) & ~(size_t)255; }

extern "C" void kernel_launch(void* const* d_in, const int* in_sizes, int n_in,
                              void* d_out, int out_size, void* d_ws, size_t ws_size,
                              hipStream_t stream) {
  const float* inputs   = (const float*)d_in[0];   // [N, D] fp32
  const float* codebook = (const float*)d_in[1];   // [K, D] fp32
  float* out = (float*)d_out;

  // output layout: [quantized N*D][indices N][commit 1][min_dist N][cls 1]
  float* q_out   = out;
  float* idxf    = out + (size_t)N_ROWS * D_DIM;
  float* commit  = idxf + N_ROWS;
  float* mind    = commit + 1;
  float* cls     = mind + N_ROWS;

  // workspace layout
  char* ws = (char*)d_ws;
  bf16*  Abf   = (bf16*)ws;   ws += align256((size_t)N_ROWS * D_DIM * 2);
  bf16*  Cbf   = (bf16*)ws;   ws += align256((size_t)K_DIM * D_DIM * 2);
  float* x2    = (float*)ws;  ws += align256((size_t)N_ROWS * 4);
  float* c2    = (float*)ws;  ws += align256((size_t)K_DIM * 4);
  int*   idxi  = (int*)ws;    ws += align256((size_t)N_ROWS * 4);
  float* rowc  = (float*)ws;  ws += align256((size_t)N_ROWS * 4);
  float* cpart = (float*)ws;  ws += align256((size_t)N_ROWS * 4);

  (void)in_sizes; (void)n_in; (void)out_size; (void)ws_size;

  hipFuncSetAttribute(reinterpret_cast<const void*>(vq_search),
                      hipFuncAttributeMaxDynamicSharedMemorySize, SMEM_MAIN);

  vq_convert_rows<<<N_ROWS, 256, 0, stream>>>(inputs, Abf, x2);
  vq_convert_rows<<<K_DIM, 256, 0, stream>>>(codebook, Cbf, c2);
  vq_search<<<N_ROWS / MB, 256, SMEM_MAIN, stream>>>(
      Abf, Cbf, x2, c2, mind, idxf, idxi, rowc);
  vq_gather<<<N_ROWS, 256, 0, stream>>>(idxi, codebook, inputs, q_out, cpart);
  vq_finalize<<<1, 256, 0, stream>>>(rowc, cpart, commit, cls);
}
